// GNNBackbone_20933670601307
// MI455X (gfx1250) — matrix operand order repaired and checked by reading, not compiled, not run on hardware
//
#include <hip/hip_runtime.h>
#include <math.h>

typedef __attribute__((ext_vector_type(16))) __bf16 v16bf;
typedef __attribute__((ext_vector_type(8)))  float  v8f;

#define DIM   256
#define SLOPE 0.2f
#define LNEPS 1e-5f
#define MT    4          // M-tiles per wave in the main WMMA GEMM

static inline int ceil_div(int a, int b) { return (a + b - 1) / b; }

__device__ inline unsigned short f32_to_bf16_rne(float f) {
    unsigned int u = __float_as_uint(f);
    unsigned int r = 0x7FFFu + ((u >> 16) & 1u);
    return (unsigned short)((u + r) >> 16);
}

// ---------------- elementwise helpers ----------------

__global__ void k_fill(float* __restrict__ p, float v, int n) {
    int i = blockIdx.x * blockDim.x + threadIdx.x;
    if (i < n) p[i] = v;
}

__global__ void k_cvt_bf16(const float* __restrict__ in, unsigned short* __restrict__ out, int n) {
    int i = blockIdx.x * blockDim.x + threadIdx.x;
    if (i < n) out[i] = f32_to_bf16_rne(in[i]);
}

// Swizzle W (256x256 f32, row-major [k][n]) into WMMA B-fragment-major bf16 layout:
//   Bf[((tn*8 + kk)*32 + lane)*8 + v] = pack(bf16 W[k][n], bf16 W[k+1][n])
//   with n = tn*16 + (lane&15), k = kk*32 + (v>=4 ? 16 : 0) + (lane>=16 ? 8 : 0) + 2(v&3)
// so each lane's 8-dword fragment is contiguous (2x b128 loads in the GEMM).
__global__ void k_pack_w_frag(const float* __restrict__ W, unsigned int* __restrict__ Bf, int total) {
    int i = blockIdx.x * blockDim.x + threadIdx.x;
    if (i >= total) return;
    int v    = i & 7;
    int lane = (i >> 3) & 31;
    int kk   = (i >> 8) & 7;
    int tn   = i >> 11;
    int n = tn * 16 + (lane & 15);
    int k = kk * 32 + ((v >> 2) ? 16 : 0) + (lane >> 4) * 8 + 2 * (v & 3);
    unsigned int lo = f32_to_bf16_rne(W[k * DIM + n]);
    unsigned int hi = f32_to_bf16_rne(W[(k + 1) * DIM + n]);
    Bf[i] = (hi << 16) | lo;
}

// ---------------- WMMA GEMM: C[N,256] = A[N,256](bf16) x B[256,256] ----------------
// Main kernel: one wave32 computes a full 64x16 strip (MT=4 tiles), NO bounds checks.
// 32 x v_wmma_f32_16x16x32_bf16 per wave; each B fragment (2x b128) reused 4x.

__global__ void k_gemm_wmma_main(const unsigned short* __restrict__ A,
                                 const unsigned int*  __restrict__ Bf,
                                 float* __restrict__ C, int mgroups) {
    // readfirstlane -> tile coords live in SGPRs: scalar branches only, EXEC untouched
    int wave = __builtin_amdgcn_readfirstlane((blockIdx.x * blockDim.x + threadIdx.x) >> 5);
    int lane = threadIdx.x & 31;
    int tn = wave / mgroups;
    if (tn >= 16) return;                         // scalar branch (grid padding only)
    int tm0 = (wave - tn * mgroups) * MT;
    int half = lane >> 4;                         // 0 | 1
    int l15  = lane & 15;

    const unsigned int* arow0 = (const unsigned int*)(A + (size_t)(tm0 * 16 + l15) * DIM);
    v8f acc[MT] = {};
    #pragma unroll
    for (int kk = 0; kk < 8; ++kk) {              // K step = 32
        union { v16bf v; unsigned int u[8]; } b;
        const uint4* bp = (const uint4*)(Bf + (((size_t)tn * 8 + kk) * 32 + lane) * 8);
        *(uint4*)&b.u[0] = bp[0];
        *(uint4*)&b.u[4] = bp[1];
        int ka = kk * 16 + half * 4;              // dword index within a 128-dword A row
        #pragma unroll
        for (int mt = 0; mt < MT; ++mt) {
            const unsigned int* arow = arow0 + (size_t)mt * 16 * (DIM / 2);
            union { v16bf v; unsigned int u[8]; } a;
            *(uint4*)&a.u[0] = *(const uint4*)(arow + ka);      // K = kk*32 + half*8 + 2i{,+1}
            *(uint4*)&a.u[4] = *(const uint4*)(arow + ka + 8);  // K = kk*32 + 16 + half*8 + ...
            acc[mt] = __builtin_amdgcn_wmma_f32_16x16x32_bf16(false, a.v, false, b.v,
                                                              (short)0, acc[mt], false, false);
        }
    }
    #pragma unroll
    for (int mt = 0; mt < MT; ++mt) {
        float* crow = C + tn * 16 + l15;
        #pragma unroll
        for (int v = 0; v < 8; ++v)
            crow[(size_t)((tm0 + mt) * 16 + v + half * 8) * DIM] = acc[mt][v];
    }
}

// Tail kernel: one 16x16 tile per wave for rows [row0, nrows). Row-tile guard is scalar.

__global__ void k_gemm_wmma_tail(const unsigned short* __restrict__ A,
                                 const unsigned int*  __restrict__ Bf,
                                 float* __restrict__ C, int row0, int nrows) {
    int wave = __builtin_amdgcn_readfirstlane((blockIdx.x * blockDim.x + threadIdx.x) >> 5);
    int lane = threadIdx.x & 31;
    int mtiles = (nrows - row0 + 15) >> 4;
    int tm = wave >> 4;
    int tn = wave & 15;
    if (tm >= mtiles) return;                     // scalar branch
    int half = lane >> 4;
    int l15  = lane & 15;

    const unsigned int* arow = (const unsigned int*)(A + (size_t)(row0 + tm * 16 + l15) * DIM);
    v8f acc = {};
    #pragma unroll
    for (int kk = 0; kk < 8; ++kk) {
        union { v16bf v; unsigned int u[8]; } a, b;
        const uint4* bp = (const uint4*)(Bf + (((size_t)tn * 8 + kk) * 32 + lane) * 8);
        *(uint4*)&b.u[0] = bp[0];
        *(uint4*)&b.u[4] = bp[1];
        int ka = kk * 16 + half * 4;
        *(uint4*)&a.u[0] = *(const uint4*)(arow + ka);
        *(uint4*)&a.u[4] = *(const uint4*)(arow + ka + 8);
        acc = __builtin_amdgcn_wmma_f32_16x16x32_bf16(false, a.v, false, b.v,
                                                      (short)0, acc, false, false);
    }
    float* crow = C + tn * 16 + l15;
    #pragma unroll
    for (int v = 0; v < 8; ++v)
        crow[(size_t)(row0 + tm * 16 + v + half * 8) * DIM] = acc[v];
}

// ---------------- attention logits: a_s[n,h] = <xh[n,h,:], att_src[h,:]> ----------------

__global__ void k_attn_scores(const float* __restrict__ xh,
                              const float* __restrict__ att_src,
                              const float* __restrict__ att_dst,
                              float* __restrict__ a_s, float* __restrict__ a_d,
                              int n_nodes, int heads, int dh) {
    int i = blockIdx.x * blockDim.x + threadIdx.x;
    if (i >= n_nodes * heads) return;
    int n = i / heads, h = i - n * heads;
    const float* xp = xh + (size_t)n * DIM + h * dh;
    const float* sp = att_src + h * dh;
    const float* dp = att_dst + h * dh;
    float ss = 0.f, sd = 0.f;
    for (int d = 0; d < dh; ++d) { ss += xp[d] * sp[d]; sd += xp[d] * dp[d]; }
    a_s[i] = ss; a_d[i] = sd;
}

// ---------------- edge passes ----------------

__device__ inline void edge_endpoints(const int* __restrict__ ei, int e, int E,
                                      int& s, int& d) {
    if (e < E) { s = ei[e]; d = ei[E + e]; }   // edge_index[0]=src, [1]=dst
    else       { s = d = e - E; }              // self-loop
}

__device__ inline float leaky(float x) { return x > 0.f ? x : SLOPE * x; }

__global__ void k_edge_max(const int* __restrict__ ei, int E, int n_nodes,
                           const float* __restrict__ a_s, const float* __restrict__ a_d,
                           float* __restrict__ amax, int heads) {
    int t = blockIdx.x * blockDim.x + threadIdx.x;
    int total = (E + n_nodes) * heads;
    if (t >= total) return;
    int e = t / heads, h = t - e * heads;
    int s, d; edge_endpoints(ei, e, E, s, d);
    float al = leaky(a_s[s * heads + h] + a_d[d * heads + h]);
    float* addr = amax + d * heads + h;
    if (al >= 0.f) atomicMax((int*)addr, __float_as_int(al));
    else           atomicMin((unsigned int*)addr, __float_as_uint(al));
}

__global__ void k_edge_expsum(const int* __restrict__ ei, int E, int n_nodes,
                              const float* __restrict__ a_s, const float* __restrict__ a_d,
                              const float* __restrict__ amax, float* __restrict__ z, int heads) {
    int t = blockIdx.x * blockDim.x + threadIdx.x;
    int total = (E + n_nodes) * heads;
    if (t >= total) return;
    int e = t / heads, h = t - e * heads;
    int s, d; edge_endpoints(ei, e, E, s, d);
    float al = leaky(a_s[s * heads + h] + a_d[d * heads + h]);
    atomicAdd(&z[d * heads + h], __expf(al - amax[d * heads + h]));
}

// 64 threads per edge, each handles 4 contiguous dims (float4 gather + 4 atomic adds)
__global__ void k_edge_scatter(const int* __restrict__ ei, int E, int n_nodes,
                               const float* __restrict__ a_s, const float* __restrict__ a_d,
                               const float* __restrict__ amax, const float* __restrict__ z,
                               const float* __restrict__ xh, float* __restrict__ acc, int heads) {
    int t = blockIdx.x * blockDim.x + threadIdx.x;
    int e = t >> 6;
    if (e >= E + n_nodes) return;
    int dim = (t & 63) * 4;
    int dh = DIM / heads;
    int h = dim / dh;
    int s, d; edge_endpoints(ei, e, E, s, d);
    float al = leaky(a_s[s * heads + h] + a_d[d * heads + h]);
    float w = __expf(al - amax[d * heads + h]) / (z[d * heads + h] + 1e-16f);
    float4 xv = *(const float4*)(xh + (size_t)s * DIM + dim);
    float* o = acc + (size_t)d * DIM + dim;
    atomicAdd(o + 0, xv.x * w);
    atomicAdd(o + 1, xv.y * w);
    atomicAdd(o + 2, xv.z * w);
    atomicAdd(o + 3, xv.w * w);
}

// ---------------- bias + LayerNorm + residual + ReLU ----------------

__global__ void k_post_ln(const float* __restrict__ acc, const float* __restrict__ bias,
                          const float* __restrict__ g, const float* __restrict__ b,
                          const float* __restrict__ resid,   // nullptr for layer 0
                          float* __restrict__ out) {
    __shared__ float red[DIM];
    int n = blockIdx.x, d = threadIdx.x;
    float v = acc[(size_t)n * DIM + d] + bias[d];
    red[d] = v;
    for (int s = DIM / 2; s > 0; s >>= 1) { __syncthreads(); if (d < s) red[d] += red[d + s]; }
    __syncthreads();
    float mu = red[0] * (1.0f / DIM);
    __syncthreads();
    float dv = v - mu;
    red[d] = dv * dv;
    for (int s = DIM / 2; s > 0; s >>= 1) { __syncthreads(); if (d < s) red[d] += red[d + s]; }
    __syncthreads();
    float var = red[0] * (1.0f / DIM);
    float y = dv * rsqrtf(var + LNEPS) * g[d] + b[d];
    if (resid) y += resid[(size_t)n * DIM + d];
    out[(size_t)n * DIM + d] = fmaxf(y, 0.0f);
}

// ---------------- driver ----------------

extern "C" void kernel_launch(void* const* d_in, const int* in_sizes, int n_in,
                              void* d_out, int out_size, void* d_ws, size_t ws_size,
                              hipStream_t stream) {
    const int n_nodes = in_sizes[0] / DIM;
    const int E = in_sizes[1] / 2;

    const float* x  = (const float*)d_in[0];
    const int*   ei = (const int*)d_in[1];
    const float* W[3]    = {(const float*)d_in[2],  (const float*)d_in[8],  (const float*)d_in[14]};
    const float* asrc[3] = {(const float*)d_in[3],  (const float*)d_in[9],  (const float*)d_in[15]};
    const float* adst[3] = {(const float*)d_in[4],  (const float*)d_in[10], (const float*)d_in[16]};
    const float* bias[3] = {(const float*)d_in[5],  (const float*)d_in[11], (const float*)d_in[17]};
    const float* lng[3]  = {(const float*)d_in[6],  (const float*)d_in[12], (const float*)d_in[18]};
    const float* lnb[3]  = {(const float*)d_in[7],  (const float*)d_in[13], (const float*)d_in[19]};
    const int heads[3] = {4, 4, 1};

    // workspace layout
    char* w = (char*)d_ws;
    const size_t szNode = (size_t)n_nodes * DIM * sizeof(float);
    float* hA   = (float*)w;                       w += szNode;
    float* hB   = (float*)w;                       w += szNode;
    float* xh   = (float*)w;                       w += szNode;
    float* acc  = (float*)w;                       w += szNode;
    unsigned short* hbf = (unsigned short*)w;      w += (size_t)n_nodes * DIM * sizeof(unsigned short);
    unsigned int*   Bf  = (unsigned int*)w;        w += (size_t)(DIM / 2) * DIM * sizeof(unsigned int);
    float* a_s  = (float*)w;                       w += (size_t)n_nodes * 4 * sizeof(float);
    float* a_d  = (float*)w;                       w += (size_t)n_nodes * 4 * sizeof(float);
    float* amax = (float*)w;                       w += (size_t)n_nodes * 4 * sizeof(float);
    float* zbuf = (float*)w;                       w += (size_t)n_nodes * 4 * sizeof(float);

    const int TB = 256;
    const int mtiles = ceil_div(n_nodes, 16);
    const int full_groups = mtiles / MT;              // guard-free 64-row strips
    const int row_tail = full_groups * MT * 16;       // first row handled by tail kernel

    const float* hin = x;
    for (int l = 0; l < 3; ++l) {
        const int H = heads[l];
        // 1) bf16 convert input + swizzle W into B-fragment layout
        k_cvt_bf16<<<ceil_div(n_nodes * DIM, TB), TB, 0, stream>>>(hin, hbf, n_nodes * DIM);
        k_pack_w_frag<<<ceil_div((DIM / 2) * DIM, TB), TB, 0, stream>>>(W[l], Bf, (DIM / 2) * DIM);
        // 2) WMMA GEMM: xh = hin @ W
        if (full_groups > 0) {
            int waves = 16 * full_groups;
            k_gemm_wmma_main<<<ceil_div(waves * 32, TB), TB, 0, stream>>>(hbf, Bf, xh, full_groups);
        }
        if (row_tail < n_nodes) {
            int waves = 16 * ceil_div(n_nodes - row_tail, 16);
            k_gemm_wmma_tail<<<ceil_div(waves * 32, TB), TB, 0, stream>>>(hbf, Bf, xh, row_tail, n_nodes);
        }
        // 3) attention logits
        k_attn_scores<<<ceil_div(n_nodes * H, TB), TB, 0, stream>>>(
            xh, asrc[l], adst[l], a_s, a_d, n_nodes, H, DIM / H);
        // 4) init segment buffers
        k_fill<<<ceil_div(n_nodes * H, TB), TB, 0, stream>>>(amax, -INFINITY, n_nodes * H);
        k_fill<<<ceil_div(n_nodes * H, TB), TB, 0, stream>>>(zbuf, 0.0f, n_nodes * H);
        k_fill<<<ceil_div(n_nodes * DIM, TB), TB, 0, stream>>>(acc, 0.0f, n_nodes * DIM);
        // 5) edge passes (E real edges + n self-loops)
        int nedge = E + n_nodes;
        k_edge_max<<<ceil_div(nedge * H, TB), TB, 0, stream>>>(ei, E, n_nodes, a_s, a_d, amax, H);
        k_edge_expsum<<<ceil_div(nedge * H, TB), TB, 0, stream>>>(ei, E, n_nodes, a_s, a_d, amax, zbuf, H);
        k_edge_scatter<<<ceil_div(nedge * 64, TB), TB, 0, stream>>>(
            ei, E, n_nodes, a_s, a_d, amax, zbuf, xh, acc, H);
        // 6) bias + LN + residual + relu
        float* hout = (l == 2) ? (float*)d_out : (l == 0 ? hA : hB);
        k_post_ln<<<n_nodes, DIM, 0, stream>>>(acc, bias[l], lng[l], lnb[l],
                                               (l > 0) ? hin : nullptr, hout);
        hin = hout;
    }
}
